// GraphTransformer_45372034515010
// MI455X (gfx1250) — matrix operand order repaired and checked by reading, not compiled, not run on hardware
//
#include <hip/hip_runtime.h>
#include <hip/hip_bf16.h>

#define N_NODES 50000
#define E_EDGES 800000
#define DMODEL  128
#define NLAYERS 3

typedef __attribute__((ext_vector_type(16))) _Float16 v16h;
typedef __attribute__((ext_vector_type(8)))  float    v8f;
typedef __attribute__((ext_vector_type(4)))  float    v4f;

union HV { v16h h; v4f q[2]; };

// ---------------- elementwise helpers ----------------
__global__ void k_f32_to_f16(const float* __restrict__ src, _Float16* __restrict__ dst, int n) {
    int i = blockIdx.x * blockDim.x + threadIdx.x;
    int stride = gridDim.x * blockDim.x;
    for (; i < n; i += stride) dst[i] = (_Float16)src[i];
}

__global__ void k_fill_i32(int* __restrict__ p, int val, int n) {
    int i = blockIdx.x * blockDim.x + threadIdx.x;
    if (i < n) p[i] = val;
}

// ---------------- CSR build (dst is layer-invariant: build once) ----------------
__global__ void k_deg(const int* __restrict__ dst, int* __restrict__ deg, int e) {
    int i = blockIdx.x * blockDim.x + threadIdx.x;
    if (i < e) atomicAdd(&deg[dst[i]], 1);
}

__global__ void k_scan(const int* __restrict__ deg, int* __restrict__ rowptr,
                       int* __restrict__ cursor, int n) {
    __shared__ int s[1024];
    __shared__ int carry;
    int tid = threadIdx.x;
    if (tid == 0) carry = 0;
    __syncthreads();
    for (int base = 0; base < n; base += 1024) {
        int idx = base + tid;
        int val = (idx < n) ? deg[idx] : 0;
        s[tid] = val;
        __syncthreads();
        for (int off = 1; off < 1024; off <<= 1) {
            int t = (tid >= off) ? s[tid - off] : 0;
            __syncthreads();
            s[tid] += t;
            __syncthreads();
        }
        int excl = s[tid] - val + carry;
        if (idx < n) { rowptr[idx] = excl; cursor[idx] = excl; }
        __syncthreads();
        if (tid == 1023) carry += s[1023];
        __syncthreads();
    }
    if (tid == 0) rowptr[n] = carry;
}

__global__ void k_fill_edges(const int* __restrict__ src, const int* __restrict__ dst,
                             int* __restrict__ cursor, int* __restrict__ srcs, int e) {
    int i = blockIdx.x * blockDim.x + threadIdx.x;
    if (i < e) {
        int d = dst[i];
        int pos = atomicAdd(&cursor[d], 1);
        srcs[pos] = src[i];
    }
}

// ---------------- WMMA f16 GEMM: C[M,128] = f32(A16[M,128] @ W16[128,128]) + bias ----------
// 256 thr = 8 waves; each wave owns a 16-row stripe; W staged transposed in LDS.
__global__ __launch_bounds__(256, 2)
void k_gemm128(const _Float16* __restrict__ A, const _Float16* __restrict__ W,
               const float* __restrict__ bias, float* __restrict__ C, int M) {
    __shared__ __align__(16) _Float16 Wt[128 * 136];   // [n][k], padded
    int tid = threadIdx.x;
    for (int i = tid; i < 128 * 128; i += 256) {
        int k = i >> 7, n = i & 127;
        Wt[n * 136 + k] = W[i];                        // W is [k][n] row-major
    }
    __syncthreads();

    int wave = tid >> 5;
    int lane = tid & 31;
    int l16  = lane & 15;
    int lhi  = lane >> 4;                              // 0 or 1
    int rowbase = blockIdx.x * 128 + wave * 16;
    int arow = rowbase + l16; if (arow >= M) arow = M - 1;

    v8f c[8];
#pragma unroll
    for (int nt = 0; nt < 8; ++nt) {
        float b = bias[nt * 16 + l16];                 // bias depends only on output col
#pragma unroll
        for (int j = 0; j < 8; ++j) c[nt][j] = b;
    }

#pragma unroll
    for (int kt = 0; kt < 4; ++kt) {                   // K = 128 = 4 x 32
        // A fragment (16x32 f16): lanes 0-15 -> K {0..7,16..23}, lanes 16-31 -> K {8..15,24..31}
        HV a;
        const v4f* ap = reinterpret_cast<const v4f*>(A + (size_t)arow * 128 + kt * 32 + lhi * 8);
        a.q[0] = ap[0];                                // 8 consecutive halves -> VGPR0..3
        a.q[1] = ap[2];                                // +16 halves -> VGPR4..7
        int kstart = kt * 32 + lhi * 8;                // B: lanes 0-15 K {0..7,16..23}, lanes 16-31 K {8..15,24..31}
#pragma unroll
        for (int nt = 0; nt < 8; ++nt) {
            HV b;
            const v4f* bp = reinterpret_cast<const v4f*>(&Wt[(nt * 16 + l16) * 136 + kstart]);
            b.q[0] = bp[0];
            b.q[1] = bp[2];
            c[nt] = __builtin_amdgcn_wmma_f32_16x16x32_f16(
                        false, a.h, false, b.h, (short)0, c[nt], false, false);
        }
    }

#pragma unroll
    for (int nt = 0; nt < 8; ++nt) {
#pragma unroll
        for (int r = 0; r < 8; ++r) {
            int row = rowbase + lhi * 8 + r;           // C layout: vgpr r = row r / r+8
            if (row < M) C[(size_t)row * 128 + nt * 16 + l16] = c[nt][r];
        }
    }
}

// ---------------- per-node attention: block=128 (wave w = head w, lane = channel) ----------
__global__ void k_attn(const float* __restrict__ q, const float* __restrict__ kk,
                       const float* __restrict__ vv, const float* __restrict__ rr,
                       const int* __restrict__ rowptr, const int* __restrict__ srcs,
                       float* __restrict__ hout, _Float16* __restrict__ h16) {
    int n    = blockIdx.x;
    int col  = threadIdx.x;                            // head = col>>5, channel = col&31
    float qv = q[(size_t)n * 128 + col];
    int beg = rowptr[n], end = rowptr[n + 1];
    float m = -3.402823466e38f, z = 0.f, acc = 0.f;
    for (int e = beg; e < end; ++e) {
        int s = srcs[e];
        float p = qv * kk[(size_t)s * 128 + col];
#pragma unroll
        for (int off = 16; off > 0; off >>= 1) p += __shfl_xor(p, off, 32);
        p *= 0.17677669529663687f;                     // 1/sqrt(32)
        float mn = fmaxf(m, p);
        float sc = __expf(m - mn);                     // online softmax rescale
        float we = __expf(p - mn);
        z   = z   * sc + we;
        acc = acc * sc + we * vv[(size_t)s * 128 + col];
        m = mn;
    }
    float o = acc / (z + 1e-16f) + rr[(size_t)n * 128 + col];  // + root/skip GEMM
    hout[(size_t)n * 128 + col] = o;
    h16[(size_t)n * 128 + col] = (_Float16)o;          // f16 copy for next layer's GEMMs
}

// ---------------- final sigmoid(h @ Wo + bo): one wave per node ----------------
__global__ void k_out(const float* __restrict__ h, const float* __restrict__ Wo,
                      const float* __restrict__ bo, float* __restrict__ out, int n) {
    int wave = threadIdx.x >> 5, lane = threadIdx.x & 31;
    int node = blockIdx.x * 8 + wave;
    if (node >= n) return;
    float s = 0.f;
#pragma unroll
    for (int i = lane; i < 128; i += 32) s += h[(size_t)node * 128 + i] * Wo[i];
#pragma unroll
    for (int off = 16; off > 0; off >>= 1) s += __shfl_xor(s, off, 32);
    if (lane == 0) out[node] = 1.f / (1.f + __expf(-(s + bo[0])));
}

extern "C" void kernel_launch(void* const* d_in, const int* in_sizes, int n_in,
                              void* d_out, int out_size, void* d_ws, size_t ws_size,
                              hipStream_t stream) {
    const float* x  = (const float*)d_in[0];
    const int*   ei = (const int*)  d_in[1];
    const float* Wi = (const float*)d_in[2];
    const float* bi = (const float*)d_in[3];
    const float* Wq = (const float*)d_in[4];
    const float* bq = (const float*)d_in[5];
    const float* Wk = (const float*)d_in[6];
    const float* bk = (const float*)d_in[7];
    const float* Wv = (const float*)d_in[8];
    const float* bv = (const float*)d_in[9];
    const float* Wr = (const float*)d_in[10];
    const float* br = (const float*)d_in[11];
    const float* Wo = (const float*)d_in[12];
    const float* bo = (const float*)d_in[13];
    float* out = (float*)d_out;

    const int* srcI = ei;
    const int* dstI = ei + E_EDGES;

    char* ws = (char*)d_ws;
    size_t off = 0;
    auto alloc = [&](size_t bytes) -> void* {
        off = (off + 255) & ~(size_t)255;
        void* p = ws + off;
        off += bytes;
        return p;
    };
    _Float16* h16    = (_Float16*)alloc((size_t)N_NODES * 128 * 2);
    _Float16* w16    = (_Float16*)alloc((size_t)13 * 16384 * 2);
    float*    q32    = (float*)   alloc((size_t)N_NODES * 128 * 4);
    float*    k32    = (float*)   alloc((size_t)N_NODES * 128 * 4);
    float*    v32    = (float*)   alloc((size_t)N_NODES * 128 * 4);
    float*    r32    = (float*)   alloc((size_t)N_NODES * 128 * 4);
    float*    h32    = (float*)   alloc((size_t)N_NODES * 128 * 4);
    int*      deg    = (int*)     alloc((size_t)N_NODES * 4);
    int*      rowptr = (int*)     alloc((size_t)(N_NODES + 1) * 4);
    int*      cursor = (int*)     alloc((size_t)N_NODES * 4);
    int*      srcs   = (int*)     alloc((size_t)E_EDGES * 4);

    _Float16* wi16 = w16;
    _Float16* wq16 = w16 + 16384;
    _Float16* wk16 = w16 + 16384 * 4;
    _Float16* wv16 = w16 + 16384 * 7;
    _Float16* wr16 = w16 + 16384 * 10;

    // f32 -> f16 staging (weights once, x once)
    k_f32_to_f16<<<(N_NODES * 128 + 255) / 256, 256, 0, stream>>>(x, h16, N_NODES * 128);
    k_f32_to_f16<<<64,  256, 0, stream>>>(Wi, wi16, 16384);
    k_f32_to_f16<<<192, 256, 0, stream>>>(Wq, wq16, 3 * 16384);
    k_f32_to_f16<<<192, 256, 0, stream>>>(Wk, wk16, 3 * 16384);
    k_f32_to_f16<<<192, 256, 0, stream>>>(Wv, wv16, 3 * 16384);
    k_f32_to_f16<<<192, 256, 0, stream>>>(Wr, wr16, 3 * 16384);

    // CSR by destination (layer-invariant)
    k_fill_i32<<<(N_NODES + 255) / 256, 256, 0, stream>>>(deg, 0, N_NODES);
    k_deg<<<(E_EDGES + 255) / 256, 256, 0, stream>>>(dstI, deg, E_EDGES);
    k_scan<<<1, 1024, 0, stream>>>(deg, rowptr, cursor, N_NODES);
    k_fill_edges<<<(E_EDGES + 255) / 256, 256, 0, stream>>>(srcI, dstI, cursor, srcs, E_EDGES);

    int gblocks = (N_NODES + 127) / 128;
    // h = x @ Wi + bi
    k_gemm128<<<gblocks, 256, 0, stream>>>(h16, wi16, bi, h32, N_NODES);
    k_f32_to_f16<<<(N_NODES * 128 + 255) / 256, 256, 0, stream>>>(h32, h16, N_NODES * 128);

    for (int l = 0; l < NLAYERS; ++l) {
        k_gemm128<<<gblocks, 256, 0, stream>>>(h16, wq16 + l * 16384, bq + l * 128, q32, N_NODES);
        k_gemm128<<<gblocks, 256, 0, stream>>>(h16, wk16 + l * 16384, bk + l * 128, k32, N_NODES);
        k_gemm128<<<gblocks, 256, 0, stream>>>(h16, wv16 + l * 16384, bv + l * 128, v32, N_NODES);
        k_gemm128<<<gblocks, 256, 0, stream>>>(h16, wr16 + l * 16384, br + l * 128, r32, N_NODES);
        k_attn<<<N_NODES, 128, 0, stream>>>(q32, k32, v32, r32, rowptr, srcs, h32, h16);
    }

    k_out<<<(N_NODES + 7) / 8, 256, 0, stream>>>(h32, Wo, bo, out, N_NODES);
}
